// NodeLevelAttentionLayer_65910568124774
// MI455X (gfx1250) — matrix operand order repaired and checked by reading, not compiled, not run on hardware
//
#include <hip/hip_runtime.h>
#include <hip/hip_bf16.h>

typedef __attribute__((ext_vector_type(16))) __bf16 v16bf;
typedef __attribute__((ext_vector_type(8)))  __bf16 v8bf;
typedef __attribute__((ext_vector_type(8)))  float  v8f;

#define N_NODES 4096
#define N_EDGES 16384
#define FN 64
#define FE 32
#define DN 128
#define DE 64
#define KC_N 8    // K-split chunks for node aggregation (4096/8 = 512 per chunk)
#define KC_E 16   // K-split chunks for edge aggregation (16384/16 = 1024 per chunk)

__device__ __forceinline__ float lrelu(float x) { return x >= 0.f ? x : 0.01f * x; }

// Workspace layout in float units (~10.5 MiB total; node/edge partials share WS_PART
// since the two aggregation passes are stream-serialized).
enum {
  WS_CSELF_N = 0,                    // 64  : weight_node @ pvn[:128]
  WS_CNBR_N  = 64,                   // 64  : weight_node @ pvn[128:]
  WS_CSELF_E = 128,                  // 64  : weight_node @ pve[:128]
  WS_CNBR_E  = 192,                  // 32  : weight_edge @ pve[128:]
  WS_S0N     = 224,
  WS_S0B     = 225,
  WS_ASELF   = 256,                  // 4096
  WS_ANBR    = WS_ASELF + 4096,
  WS_BSELF   = WS_ANBR + 4096,
  WS_ESELF_N = WS_BSELF + 4096,      // exp(leaky(a_self))
  WS_ESELF_B = WS_ESELF_N + 4096,    // exp(leaky(b_self))
  WS_BNBR    = WS_ESELF_B + 4096,    // 16384
  WS_DINV_N  = WS_BNBR + 16384,      // 4096
  WS_DINV_E  = WS_DINV_N + 4096,     // 16384 -> ends 57600
  WS_PART    = 57600,                // 2,097,152 f32 (= 8*4096*64 = 16*4096*32), shared
  WS_MEANS_NB = WS_PART + 2097152,   // 4096*64 bf16  (131072 f32 slots)
  WS_MEANS_EB = WS_MEANS_NB + 131072,// 4096*32 bf16  (65536 slots)
  WS_NFT      = WS_MEANS_EB + 65536, // node feats^T bf16 [64][4096]  (131072 slots)
  WS_EFT      = WS_NFT + 131072,     // edge feats^T bf16 [32][16384] (262144 slots)
  WS_WNT      = WS_EFT + 262144,     // weight_node^T bf16 [128][64]  (4096 slots)
  WS_WET      = WS_WNT + 4096,       // weight_edge^T bf16 [64][32]   (1024 slots)
  WS_TOTAL    = WS_WET + 1024        // = 2,749,696 floats ~= 10.5 MiB
};

// ---- fold parameter vectors through the weight matrices -------------------
__global__ void k_prep(const float* __restrict__ wn, const float* __restrict__ we,
                       const float* __restrict__ pvn, const float* __restrict__ pve,
                       float* __restrict__ ws) {
  int t = threadIdx.x;
  if (t < 64) {
    float s0 = 0.f, s1 = 0.f, s2 = 0.f;
    for (int d = 0; d < DN; ++d) {
      float w = wn[t * DN + d];
      s0 += w * pvn[d];
      s1 += w * pvn[DN + d];
      s2 += w * pve[d];
    }
    ws[WS_CSELF_N + t] = s0;
    ws[WS_CNBR_N + t]  = s1;
    ws[WS_CSELF_E + t] = s2;
  } else if (t < 96) {
    int r = t - 64;
    float s = 0.f;
    for (int d = 0; d < DE; ++d) s += we[r * DE + d] * pve[DN + d];
    ws[WS_CNBR_E + r] = s;
  }
}

// ---- transpose + convert to bf16 (features, weights) ----------------------
__global__ void k_tobf16t(const float* __restrict__ src, int R, int C,
                          __bf16* __restrict__ dst) {
  int idx = blockIdx.x * blockDim.x + threadIdx.x;
  if (idx >= R * C) return;
  int r = idx / C, c = idx % C;
  dst[(size_t)c * R + r] = (__bf16)src[idx];
}

// ---- per-row attention scalars --------------------------------------------
__global__ void k_proj(const float* __restrict__ nf, const float* __restrict__ ef,
                       float* __restrict__ ws) {
  int i = blockIdx.x * blockDim.x + threadIdx.x;
  if (i < N_NODES) {
    float s0 = 0.f, s1 = 0.f, s2 = 0.f;
    for (int k = 0; k < FN; ++k) {
      float v = nf[(size_t)i * FN + k];
      s0 += v * ws[WS_CSELF_N + k];
      s1 += v * ws[WS_CNBR_N + k];
      s2 += v * ws[WS_CSELF_E + k];
    }
    ws[WS_ASELF + i]   = s0;
    ws[WS_ANBR + i]    = s1;
    ws[WS_BSELF + i]   = s2;
    ws[WS_ESELF_N + i] = __expf(lrelu(s0));
    ws[WS_ESELF_B + i] = __expf(lrelu(s2));
  }
  if (i < N_EDGES) {
    float s = 0.f;
    for (int k = 0; k < FE; ++k) s += ef[(size_t)i * FE + k] * ws[WS_CNBR_E + k];
    ws[WS_BNBR + i] = s;
  }
}

// ---- global sums S0 = sum_i exp(leaky(self_i)) ----------------------------
__global__ void k_s0(float* __restrict__ ws) {
  __shared__ float ra[256], rb[256];
  int t = threadIdx.x;
  float sa = 0.f, sb = 0.f;
  for (int i = t; i < N_NODES; i += 256) {
    sa += ws[WS_ESELF_N + i];
    sb += ws[WS_ESELF_B + i];
  }
  ra[t] = sa; rb[t] = sb;
  __syncthreads();
  for (int s = 128; s > 0; s >>= 1) {
    if (t < s) { ra[t] += ra[t + s]; rb[t] += rb[t + s]; }
    __syncthreads();
  }
  if (t == 0) { ws[WS_S0N] = ra[0]; ws[WS_S0B] = rb[0]; }
}

// ---- softmax column denominators (one block per mask row, streaming) ------
__global__ void k_denom(const float* __restrict__ adj, const float* __restrict__ a_self,
                        const float* __restrict__ a_nbr_j, const float* __restrict__ e_self,
                        const float* __restrict__ s0ptr, float* __restrict__ dinv) {
  int j = blockIdx.x;
  float anj = a_nbr_j[j];
  const float* row = adj + (size_t)j * N_NODES;
  float acc = 0.f;
  for (int i = threadIdx.x; i < N_NODES; i += blockDim.x) {
    float m = row[i];
    if (m != 0.f) acc += __expf(lrelu(a_self[i] + anj)) - e_self[i];
  }
  __shared__ float red[256];
  red[threadIdx.x] = acc;
  __syncthreads();
  for (int s = 128; s > 0; s >>= 1) {
    if (threadIdx.x < s) red[threadIdx.x] += red[threadIdx.x + s];
    __syncthreads();
  }
  if (threadIdx.x == 0) dinv[j] = 1.f / (s0ptr[0] + red[0]);
}

// ---- fused attention-weight x features GEMM via bf16 WMMA -----------------
// partial[chunk][i][f] = sum_{j in chunk} w[i,j]*feat[j,f]
// w[i,j] = (adj[j,i]!=0) ? exp(leaky(a_self[i]+a_nbr[j]))*dinv[j] : 0
// One wave per block; two 16-row M-tiles (32 cols of adj = one full 128B line);
// A fragments built once per K=32 step, B fragment (bf16, contiguous via the
// transposed feature copy) reused across both M-tiles.
template <int FW>
__global__ void k_agg(const float* __restrict__ adj, const __bf16* __restrict__ featT,
                      const float* __restrict__ a_self, const float* __restrict__ a_nbr,
                      const float* __restrict__ dinv, float* __restrict__ part,
                      int kdim, int kchunk) {
  constexpr int FT = FW / 16;
  int lane = threadIdx.x;
  int m  = lane & 15;
  int hi = lane >> 4;
  int i0 = blockIdx.x * 32;
  int jbeg = blockIdx.y * kchunk;
  int jend = jbeg + kchunk;
  float asl0 = a_self[i0 + m];
  float asl1 = a_self[i0 + 16 + m];

  v8f c[2][FT];
#pragma unroll
  for (int mt = 0; mt < 2; ++mt)
#pragma unroll
    for (int t = 0; t < FT; ++t)
      c[mt][t] = (v8f){0.f, 0.f, 0.f, 0.f, 0.f, 0.f, 0.f, 0.f};

  for (int j0 = jbeg; j0 < jend; j0 += 32) {
    if (j0 + 32 < jend)  // prefetch next 32-row mask slab (one line per lane)
      __builtin_prefetch(adj + (size_t)(j0 + 32 + lane) * N_NODES + i0, 0, 3);

    // A fragments: this code's contiguous k order (B uses the same).
    // lanes 0-15: K=0..15; lanes 16-31: K=16..31
    v16bf a0, a1;
#pragma unroll
    for (int e = 0; e < 16; ++e) {
      int k = hi * 16 + e;
      int j = j0 + k;
      const float* arow = adj + (size_t)j * N_NODES + i0 + m;
      float mv0 = arow[0];
      float mv1 = arow[16];
      float w0 = 0.f, w1 = 0.f;
      if ((mv0 != 0.f) || (mv1 != 0.f)) {
        float anb = a_nbr[j];
        float di  = dinv[j];
        if (mv0 != 0.f) w0 = __expf(lrelu(asl0 + anb)) * di;
        if (mv1 != 0.f) w1 = __expf(lrelu(asl1 + anb)) * di;
      }
      a0[e] = (__bf16)w0;
      a1[e] = (__bf16)w1;
    }
#pragma unroll
    for (int t = 0; t < FT; ++t) {
      // B fragment: 32x16, lanes 0-15 K=0..15, lanes 16-31 K=16..31, N=lane&15.
      // K-values contiguous in the transposed bf16 copy -> one 32B vector load.
      v16bf b = *(const v16bf*)(featT + (size_t)(t * 16 + m) * kdim + j0 + hi * 16);
      c[0][t] = __builtin_amdgcn_wmma_f32_16x16x32_bf16(false, a0, false, b, (short)0,
                                                        c[0][t], false, false);
      c[1][t] = __builtin_amdgcn_wmma_f32_16x16x32_bf16(false, a1, false, b, (short)0,
                                                        c[1][t], false, false);
    }
  }
  // C/D layout: VGPR r -> M = r + hi*8, N = lane&15
  size_t base = (size_t)blockIdx.y * N_NODES + i0;
#pragma unroll
  for (int mt = 0; mt < 2; ++mt)
#pragma unroll
    for (int t = 0; t < FT; ++t)
#pragma unroll
      for (int r = 0; r < 8; ++r) {
        int mo = mt * 16 + r + hi * 8;
        part[(base + mo) * FW + t * 16 + m] = c[mt][t][r];
      }
}

// ---- reduce K-chunk partials, scale, convert means to bf16 ----------------
__global__ void k_reduce(const float* __restrict__ part, int KC, int FW, float scale,
                         __bf16* __restrict__ outbf) {
  int idx = blockIdx.x * blockDim.x + threadIdx.x;
  if (idx >= N_NODES * FW) return;
  int i = idx / FW, f = idx % FW;
  float s = 0.f;
  for (int c = 0; c < KC; ++c) s += part[((size_t)c * N_NODES + i) * FW + f];
  outbf[idx] = (__bf16)(s * scale);
}

// ---- final projection + leaky into concatenated output --------------------
// A = means (bf16 row-major, K contiguous -> two 16B loads per fragment)
// Bt = weight^T (bf16, [Nw][K], K contiguous -> one 32B load per fragment)
__global__ void k_outgemm(const __bf16* __restrict__ A, int K, const __bf16* __restrict__ Bt,
                          float* __restrict__ out, int colbase) {
  int lane = threadIdx.x;
  int m  = lane & 15;
  int hi = lane >> 4;
  int i0 = blockIdx.x * 16;
  int f0 = threadIdx.y * 16;
  v8f c = (v8f){0.f, 0.f, 0.f, 0.f, 0.f, 0.f, 0.f, 0.f};
  for (int k0 = 0; k0 < K; k0 += 32) {
    const __bf16* arow = A + (size_t)(i0 + m) * K + k0;
    v8bf alo = *(const v8bf*)(arow + hi * 16);
    v8bf ahi = *(const v8bf*)(arow + hi * 16 + 8);
    v16bf a;
#pragma unroll
    for (int e = 0; e < 8; ++e) { a[e] = alo[e]; a[e + 8] = ahi[e]; }
    v16bf b = *(const v16bf*)(Bt + (size_t)(f0 + m) * K + k0 + hi * 16);
    c = __builtin_amdgcn_wmma_f32_16x16x32_bf16(false, a, false, b, (short)0, c,
                                                false, false);
  }
#pragma unroll
  for (int r = 0; r < 8; ++r) {
    int mo = r + hi * 8;
    out[(size_t)(i0 + mo) * (DN + DE) + colbase + f0 + m] = lrelu(c[r]);
  }
}

extern "C" void kernel_launch(void* const* d_in, const int* in_sizes, int n_in,
                              void* d_out, int out_size, void* d_ws, size_t ws_size,
                              hipStream_t stream) {
  (void)in_sizes; (void)n_in; (void)out_size; (void)ws_size; // needs ~11 MB of ws
  const float* nf  = (const float*)d_in[0];  // [4096,64]
  const float* ef  = (const float*)d_in[1];  // [16384,32]
  const float* n2n = (const float*)d_in[2];  // [4096,4096]
  const float* e2n = (const float*)d_in[3];  // [16384,4096]
  const float* wn  = (const float*)d_in[4];  // [64,128]
  const float* we  = (const float*)d_in[5];  // [32,64]
  const float* pvn = (const float*)d_in[6];  // [256]
  const float* pve = (const float*)d_in[7];  // [192]
  float* out = (float*)d_out;                // [4096,192]
  float* ws  = (float*)d_ws;

  __bf16* nft = (__bf16*)(ws + WS_NFT);
  __bf16* eft = (__bf16*)(ws + WS_EFT);
  __bf16* wnt = (__bf16*)(ws + WS_WNT);
  __bf16* wet = (__bf16*)(ws + WS_WET);
  __bf16* mnb = (__bf16*)(ws + WS_MEANS_NB);
  __bf16* meb = (__bf16*)(ws + WS_MEANS_EB);

  k_prep<<<1, 128, 0, stream>>>(wn, we, pvn, pve, ws);
  k_tobf16t<<<(N_NODES * FN) / 256, 256, 0, stream>>>(nf, N_NODES, FN, nft);
  k_tobf16t<<<(N_EDGES * FE) / 256, 256, 0, stream>>>(ef, N_EDGES, FE, eft);
  k_tobf16t<<<(FN * DN) / 256, 256, 0, stream>>>(wn, FN, DN, wnt);
  k_tobf16t<<<(FE * DE) / 256, 256, 0, stream>>>(we, FE, DE, wet);
  k_proj<<<N_EDGES / 256, 256, 0, stream>>>(nf, ef, ws);
  k_s0<<<1, 256, 0, stream>>>(ws);
  k_denom<<<N_NODES, 256, 0, stream>>>(n2n, ws + WS_ASELF, ws + WS_ANBR,
                                       ws + WS_ESELF_N, ws + WS_S0N, ws + WS_DINV_N);
  k_denom<<<N_EDGES, 256, 0, stream>>>(e2n, ws + WS_BSELF, ws + WS_BNBR,
                                       ws + WS_ESELF_B, ws + WS_S0B, ws + WS_DINV_E);
  // node aggregation (uses shared partial buffer, then reduce)
  k_agg<FN><<<dim3(N_NODES / 32, KC_N), 32, 0, stream>>>(
      n2n, nft, ws + WS_ASELF, ws + WS_ANBR, ws + WS_DINV_N, ws + WS_PART,
      N_NODES, N_NODES / KC_N);
  k_reduce<<<(N_NODES * FN) / 256, 256, 0, stream>>>(ws + WS_PART, KC_N, FN,
                                                     1.f / N_NODES, mnb);
  // edge aggregation (reuses the same partial buffer; stream-ordered)
  k_agg<FE><<<dim3(N_NODES / 32, KC_E), 32, 0, stream>>>(
      e2n, eft, ws + WS_BSELF, ws + WS_BNBR, ws + WS_DINV_E, ws + WS_PART,
      N_EDGES, N_EDGES / KC_E);
  k_reduce<<<(N_NODES * FE) / 256, 256, 0, stream>>>(ws + WS_PART, KC_E, FE,
                                                     1.f / N_EDGES, meb);
  // final projections into concatenated [4096, 192] output
  k_outgemm<<<N_NODES / 16, dim3(32, 8), 0, stream>>>(mnb, FN, wnt, out, 0);
  k_outgemm<<<N_NODES / 16, dim3(32, 4), 0, stream>>>(meb, FE, wet, out, DN);
}
